// ElementGNN_43413529428753
// MI455X (gfx1250) — matrix operand order repaired and checked by reading, not compiled, not run on hardware
//
#include <hip/hip_runtime.h>
#include <hip/hip_bf16.h>

typedef __attribute__((ext_vector_type(16))) _Float16 v16h;
typedef __attribute__((ext_vector_type(8)))  _Float16 v8h;
typedef __attribute__((ext_vector_type(8)))  float    v8f;
typedef __attribute__((ext_vector_type(4)))  float    v4f;

// ---------------------------------------------------------------------------
// Weight fragment layout in d_ws (f16):
//   per (layer, nt, kc): one 512-f16 block = 32 lanes x 16 halves.
//   B-fragment element (lane, j):  K = kc*32 + (lane/16)*8 + (j%8) + (j/8)*16
//                                  N = nt*16 + (lane%16)
//   block index = layerOff + nt*KCH + kc
// Bias folding: layers 0/5/6 carry their bias as weight row K=Din against a
// constant-1.0 activation element (K=10 for L0, K=8 for L5/L6).
// Layer offsets (in 512-f16 blocks):
//   L0(10x128,KCH=1,NT=8):0   L1(128x128,4,8):8   L2:40   L3:72
//   L4(128x8,4,1):104         L5(8x8,1,1):108     L6(8x10,1,1):109  -> 110 blocks
// ---------------------------------------------------------------------------
#define EG_NBLK 110

__global__ __launch_bounds__(256) void eg_convert_weights(
    const float* __restrict__ W0, const float* __restrict__ W1,
    const float* __restrict__ W2, const float* __restrict__ W3,
    const float* __restrict__ W4, const float* __restrict__ W5,
    const float* __restrict__ W6,
    const float* __restrict__ b0, const float* __restrict__ b5,
    const float* __restrict__ b6, _Float16* __restrict__ wfrag)
{
    const float* Ws[7] = {W0, W1, W2, W3, W4, W5, W6};
    const float* Bs[7] = {b0, 0, 0, 0, 0, b5, b6};
    const int Din[7]   = {10, 128, 128, 128, 128, 8, 8};
    const int Dout[7]  = {128, 128, 128, 128, 8, 8, 10};
    const int brow[7]  = {10, -1, -1, -1, -1, 8, 8};   // bias row (== Din)
    const int off[7]   = {0, 8, 40, 72, 104, 108, 109};
    const int kch[7]   = {1, 4, 4, 4, 4, 1, 1};

    for (int idx = threadIdx.x; idx < EG_NBLK * 512; idx += blockDim.x) {
        int blk    = idx >> 9;
        int within = idx & 511;
        int lane   = within >> 4;
        int j      = within & 15;
        int l = 0;
        #pragma unroll
        for (int t = 1; t < 7; ++t) if (blk >= off[t]) l = t;
        int local = blk - off[l];
        int kc = local % kch[l];
        int nt = local / kch[l];
        int K = kc * 32 + ((lane >> 4) << 3) + (j & 7) + ((j >> 3) << 4);
        int N = nt * 16 + (lane & 15);
        float v = 0.0f;
        if (N < Dout[l]) {
            if (K < Din[l])        v = Ws[l][K * Dout[l] + N];
            else if (K == brow[l]) v = Bs[l][N];
        }
        wfrag[idx] = (_Float16)v;
    }
}

// ---------------------------------------------------------------------------
// Fragment helpers
// ---------------------------------------------------------------------------
__device__ __forceinline__ v16h eg_loadB(const _Float16* __restrict__ wfrag,
                                         int blk, int lane)
{
    const v8h* p = (const v8h*)(wfrag + (blk << 9) + (lane << 4));
    union { v16h v; v8h h[2]; } u;
    u.h[0] = p[0];
    u.h[1] = p[1];
    return u.v;
}

// A fragment (16x32 f16) from LDS activation row (row-major, stride 136 halves).
// Lane layout per ISA: elems 0..7  -> K = kc*32 + kHalf + {0..7}
//                      elems 8..15 -> K = kc*32 + kHalf + 16 + {0..7}
__device__ __forceinline__ v16h eg_loadA(const _Float16* actRow, int kc, int kHalf)
{
    const _Float16* r = actRow + kc * 32 + kHalf;
    union { v16h v; v8h h[2]; } u;
    u.h[0] = *(const v8h*)(r);
    u.h[1] = *(const v8h*)(r + 16);
    return u.v;
}

// A fragment whose upper K half (K>=16) is all zero (for the 8/10-wide layers).
__device__ __forceinline__ v16h eg_loadA16(const _Float16* actRow, int kHalf)
{
    union { v16h v; v8h h[2]; } u;
    u.h[0] = *(const v8h*)(actRow + kHalf);
    #pragma unroll
    for (int j = 0; j < 8; ++j) u.h[1][j] = (_Float16)0.0f;
    return u.v;
}

// Single-instruction relu: v_med3_f32(x, 0, +inf) -- avoids canonicalize+max.
__device__ __forceinline__ float eg_relu(float x)
{
    return __builtin_amdgcn_fmed3f(x, 0.0f, __builtin_huge_valf());
}

// Dense 128-K layer over TWO 16-row M-tiles sharing every B fragment.
// If padOne, writes 1.0 at output column n == dimOut (bias hook for next layer).
__device__ __forceinline__ void eg_layer128x2(
    _Float16* actw, const _Float16* actRow0, const _Float16* actRow1,
    const _Float16* __restrict__ wfrag, int blkOff, int NT,
    const float* __restrict__ bias, int dimOut, bool relu, bool padOne, int lane)
{
    const int nCol  = lane & 15;
    const int mBase = (lane >> 4) * 8;
    const int kHalf = (lane >> 4) * 8;

    v16h a0[4], a1[4];
    #pragma unroll
    for (int kc = 0; kc < 4; ++kc) {
        a0[kc] = eg_loadA(actRow0, kc, kHalf);
        a1[kc] = eg_loadA(actRow1, kc, kHalf);
    }

    for (int nt = 0; nt < NT; ++nt) {
        int n = nt * 16 + nCol;
        float bv = (n < dimOut) ? bias[n] : 0.0f;
        v8f c0 = {bv, bv, bv, bv, bv, bv, bv, bv};
        v8f c1 = c0;
        #pragma unroll
        for (int kc = 0; kc < 4; ++kc) {
            v16h b = eg_loadB(wfrag, blkOff + nt * 4 + kc, lane);
            c0 = __builtin_amdgcn_wmma_f32_16x16x32_f16(
                     false, a0[kc], false, b, (short)0, c0, false, false);
            c1 = __builtin_amdgcn_wmma_f32_16x16x32_f16(
                     false, a1[kc], false, b, (short)0, c1, false, false);
        }
        #pragma unroll
        for (int r = 0; r < 8; ++r) {
            float x0 = c0[r], x1 = c1[r];
            if (relu) { x0 = eg_relu(x0); x1 = eg_relu(x1); }
            if (padOne && n == dimOut) { x0 = 1.0f; x1 = 1.0f; }
            actw[(mBase + r) * 136 + n]      = (_Float16)x0;
            actw[(16 + mBase + r) * 136 + n] = (_Float16)x1;
        }
    }
}

// ---------------------------------------------------------------------------
// Main kernel: 4 waves/block, 32 rows (two 16-row M-tiles) per wave.
// All LDS is per-wave; in-order DS hardware ordering replaces barriers.
// ---------------------------------------------------------------------------
__global__ __launch_bounds__(128) void eg_main(
    const float* __restrict__ data, const int* __restrict__ elements,
    const float* __restrict__ b1, const float* __restrict__ b2,
    const float* __restrict__ b3, const float* __restrict__ b4,
    const _Float16* __restrict__ wfrag,
    float* __restrict__ out, int nElem)
{
    __shared__ __align__(16) _Float16 s_act[4][32 * 136];   // 34816 B
    __shared__ __align__(16) float    s_sent[4][32 * 16];   //  8192 B

    const int w        = threadIdx.x >> 5;
    const int lane     = threadIdx.x & 31;
    const int tileBase = blockIdx.x * 128 + w * 32;

    _Float16* actw  = &s_act[w][0];
    float*    sentw = &s_sent[w][0];

    const int mRow  = lane & 15;
    const int nCol  = lane & 15;
    const int kHalf = (lane >> 4) * 8;
    const int mBase = (lane >> 4) * 8;
    const _Float16* actRow0 = actw + mRow * 136;
    const _Float16* actRow1 = actw + (16 + mRow) * 136;

    // ---- gather halo-augmented input rows into LDS (32 rows x 16 f32) ----
    #pragma unroll
    for (int t = 0; t < 16; ++t) {
        int s = t * 32 + lane;
        int m = s >> 4, c = s & 15;
        float v = 0.0f;
        if (c < 10) {
            int i = tileBase + m;
            if (i >= nElem) i = nElem - 1;
            int eid = elements[i] / 9;     // element index
            int src;
            if (c == 0)      src = ((eid == 0) ? nElem - 1 : eid - 1) * 8 + 7;
            else if (c == 9) src = ((eid == nElem - 1) ? 0 : eid + 1) * 8;
            else             src = eid * 8 + (c - 1);
            v = data[src];
        }
        sentw[s] = v;
    }

    // ---- A0 fragments from sent (f32 -> f16), K padded 10 -> 32;
    //      inject 1.0 at K=10 (bias row hook; K=10 lives in lanes>=16, elem 2)
    v16h a00, a01;
    #pragma unroll
    for (int mt = 0; mt < 2; ++mt) {
        const float* sr = sentw + (mt * 16 + mRow) * 16 + kHalf;
        v4f p0 = *(const v4f*)(sr);
        v4f p1 = *(const v4f*)(sr + 4);
        v16h t;
        #pragma unroll
        for (int j = 0; j < 4; ++j) { t[j] = (_Float16)p0[j]; t[4 + j] = (_Float16)p1[j]; }
        #pragma unroll
        for (int j = 8; j < 16; ++j) t[j] = (_Float16)0.0f;
        if (lane >= 16) t[2] = (_Float16)1.0f;          // K = 8 + 2 = 10
        if (mt == 0) a00 = t; else a01 = t;
    }

    // ---- layer 0: 10 -> 128, relu, bias folded into weight row K=10 ----
    for (int nt = 0; nt < 8; ++nt) {
        v8f c0 = {};                                     // inline-0 accumulator
        v8f c1 = {};
        v16h b = eg_loadB(wfrag, nt, lane);              // blocks 0..7
        c0 = __builtin_amdgcn_wmma_f32_16x16x32_f16(
                 false, a00, false, b, (short)0, c0, false, false);
        c1 = __builtin_amdgcn_wmma_f32_16x16x32_f16(
                 false, a01, false, b, (short)0, c1, false, false);
        #pragma unroll
        for (int r = 0; r < 8; ++r) {
            actw[(mBase + r) * 136 + nt * 16 + nCol]      = (_Float16)eg_relu(c0[r]);
            actw[(16 + mBase + r) * 136 + nt * 16 + nCol] = (_Float16)eg_relu(c1[r]);
        }
    }

    // ---- layers 1..4 ----
    eg_layer128x2(actw, actRow0, actRow1, wfrag,   8, 8, b1, 128, false, false, lane);
    eg_layer128x2(actw, actRow0, actRow1, wfrag,  40, 8, b2, 128, true,  false, lane);
    eg_layer128x2(actw, actRow0, actRow1, wfrag,  72, 8, b3, 128, false, false, lane);
    // 128->8, relu; writes 1.0 into col 8 so L5 picks up b5 via weight row K=8
    eg_layer128x2(actw, actRow0, actRow1, wfrag, 104, 1, b4,   8, true,  true,  lane);

    // ---- layer 5: 8 -> 8 (bias via K=8 row); writes 1.0 into col 8 for L6 ----
    {
        v16h a5_0 = eg_loadA16(actRow0, kHalf);
        v16h a5_1 = eg_loadA16(actRow1, kHalf);
        v8f c0 = {};
        v8f c1 = {};
        v16h b = eg_loadB(wfrag, 108, lane);
        c0 = __builtin_amdgcn_wmma_f32_16x16x32_f16(
                 false, a5_0, false, b, (short)0, c0, false, false);
        c1 = __builtin_amdgcn_wmma_f32_16x16x32_f16(
                 false, a5_1, false, b, (short)0, c1, false, false);
        #pragma unroll
        for (int r = 0; r < 8; ++r) {
            float x0 = (nCol == 8) ? 1.0f : (float)c0[r];
            float x1 = (nCol == 8) ? 1.0f : (float)c1[r];
            actw[(mBase + r) * 136 + nCol]      = (_Float16)x0;
            actw[(16 + mBase + r) * 136 + nCol] = (_Float16)x1;
        }
    }

    // ---- layer 6: 8 -> 10 (bias via K=8 row), residual + store ----
    {
        v16h a6_0 = eg_loadA16(actRow0, kHalf);
        v16h a6_1 = eg_loadA16(actRow1, kHalf);
        v8f c0 = {};
        v8f c1 = {};
        v16h b = eg_loadB(wfrag, 109, lane);
        c0 = __builtin_amdgcn_wmma_f32_16x16x32_f16(
                 false, a6_0, false, b, (short)0, c0, false, false);
        c1 = __builtin_amdgcn_wmma_f32_16x16x32_f16(
                 false, a6_1, false, b, (short)0, c1, false, false);

        if (nCol >= 1 && nCol <= 8) {
            #pragma unroll
            for (int r = 0; r < 8; ++r) {
                int M0 = mBase + r;
                int M1 = 16 + mBase + r;
                int i0 = tileBase + M0;
                int i1 = tileBase + M1;
                if (i0 < nElem)
                    out[(size_t)i0 * 8 + (nCol - 1)] = sentw[M0 * 16 + nCol] + c0[r];
                if (i1 < nElem)
                    out[(size_t)i1 * 8 + (nCol - 1)] = sentw[M1 * 16 + nCol] + c1[r];
            }
        }
    }
}

// ---------------------------------------------------------------------------
extern "C" void kernel_launch(void* const* d_in, const int* in_sizes, int n_in,
                              void* d_out, int out_size, void* d_ws, size_t ws_size,
                              hipStream_t stream)
{
    const float* data     = (const float*)d_in[0];
    const int*   elements = (const int*)d_in[1];
    const float* W[7];
    const float* b[7];
    for (int i = 0; i < 7; ++i) {
        W[i] = (const float*)d_in[2 + 2 * i];
        b[i] = (const float*)d_in[3 + 2 * i];
    }
    _Float16* wfrag = (_Float16*)d_ws;   // needs 110*512*2 = 112640 bytes
    const int nElem = in_sizes[1];       // 400000

    eg_convert_weights<<<1, 256, 0, stream>>>(
        W[0], W[1], W[2], W[3], W[4], W[5], W[6],
        b[0], b[5], b[6], wfrag);

    const int nBlocks = (nElem + 127) / 128;   // 3125 for 400000
    eg_main<<<nBlocks, 128, 0, stream>>>(
        data, elements, b[1], b[2], b[3], b[4],
        wfrag, (float*)d_out, nElem);
}
